// LRU_9844065042808
// MI455X (gfx1250) — compile-verified, matrix operand order checked by reading, not run on hardware
//
#include <hip/hip_runtime.h>
#include <stdint.h>

#define BATCH  16384
#define DIN    1024
#define DSTATE 2048
#define DOUT   1024

typedef __attribute__((ext_vector_type(16))) __bf16       v16bf;
typedef __attribute__((ext_vector_type(8)))  float        v8f;
typedef __attribute__((ext_vector_type(4)))  unsigned int u32x4;
typedef __attribute__((ext_vector_type(4)))  float        f32x4;

constexpr int BM    = 128;          // macro-tile rows (M)
constexpr int BN    = 64;           // macro-tile cols (N)
constexpr int BK    = 32;           // K chunk (one bf16 WMMA depth)
constexpr int P32   = BK / 2 + 4;   // 20 dwords per LDS row: 16 data + 4 pad (16B-aligned rows, conflict-free)
constexpr int PIT16 = 2 * P32;      // 40 halfwords per LDS row
constexpr int ROWB  = P32 * 4;      // 80 bytes per LDS row

union Frag {
    v16bf v;
    u32x4 q[2];
};

__device__ __forceinline__ unsigned short f2bf(float x) {
    unsigned b = __float_as_uint(x);
    unsigned r = b + 0x7FFFu + ((b >> 16) & 1u);   // round-to-nearest-even
    return (unsigned short)(r >> 16);
}
__device__ __forceinline__ float bf2f(unsigned short h) {
    return __uint_as_float(((unsigned)h) << 16);
}

// ---------------------------------------------------------------------------
// Async DMA of a rows x 32-bf16 tile (K contiguous in global) into a padded
// LDS tile, 16B per lane, no VGPR data round-trip.  Tracked by ASYNCcnt.
// ---------------------------------------------------------------------------
__device__ __forceinline__ void async_tile(const unsigned short* __restrict__ gsrc, int ldgE,
                                           const unsigned* lds_tile, int rows, int tid) {
    unsigned ldsb = (unsigned)(uintptr_t)lds_tile;
    const int nseg = rows * 4;                      // 4 x 16B segments per 64B row
    for (int s = tid; s < nseg; s += 256) {
        int r = s >> 2, sg = s & 3;
        unsigned lds = ldsb + (unsigned)(r * ROWB + sg * 16);
        unsigned long long g = (unsigned long long)(uintptr_t)gsrc
                             + (unsigned long long)r * ((unsigned long long)ldgE * 2)
                             + (unsigned)(sg * 16);
        asm volatile("global_load_async_to_lds_b128 %0, %1, off"
                     :: "v"(lds), "v"(g) : "memory");
    }
}
__device__ __forceinline__ void wait_async0() {
    asm volatile("s_wait_asynccnt 0x0" ::: "memory");
}

// ---------------------------------------------------------------------------
// VALU staging (used only by the small weight-fusion GEMM, which needs an
// on-the-fly transpose + per-k scaling that a DMA engine cannot do).
// ---------------------------------------------------------------------------
__device__ __forceinline__ void stage_tile(const float* __restrict__ g, int ldg, int rows,
                                           unsigned short* hi, unsigned short* lo,
                                           const float* __restrict__ scale, float sgn, int tid) {
    for (int e = tid; e < rows * BK; e += 256) {
        int r = e >> 5;
        int k = e & (BK - 1);
        float x = g[(size_t)r * ldg + k];
        if (scale) x *= sgn * scale[k];
        unsigned short h = f2bf(x);
        hi[r * PIT16 + k] = h;
        lo[r * PIT16 + k] = f2bf(x - bf2f(h));
    }
}

__device__ __forceinline__ void stage_tile_T(const float* __restrict__ g, int ldg,
                                             unsigned short* hi, unsigned short* lo, int tid) {
    for (int e = tid; e < BK * BN; e += 256) {
        int k = e >> 6;
        int n = e & (BN - 1);
        float x = g[(size_t)k * ldg + n];
        unsigned short h = f2bf(x);
        hi[n * PIT16 + k] = h;
        lo[n * PIT16 + k] = f2bf(x - bf2f(h));
    }
}

// ---------------------------------------------------------------------------
// Per-wave core: 2x2 tiles of 16x16, one BK=32 chunk, bf16x3 (~fp32) WMMA.
// 16-bit A/B layout (ISA 7.12.2): row/col = lane&15, K-half = lane>>4,
// dword index (v<4 ? v : v+4) + 4*half -> two ds_load_b128 per fragment.
// ---------------------------------------------------------------------------
__device__ __forceinline__ void wave_mma(const unsigned* As_hi, const unsigned* As_lo,
                                         const unsigned* Bs_hi, const unsigned* Bs_lo,
                                         v8f acc[2][2], int waveM, int waveN, int lane) {
    const int half = lane >> 4;
    const int rl   = lane & 15;
    Frag aH[2], aL[2], bH[2], bL[2];
#pragma unroll
    for (int t = 0; t < 2; ++t) {
        int row = waveM * 32 + t * 16 + rl;
        const u32x4* qh = (const u32x4*)(As_hi + row * P32);
        const u32x4* ql = (const u32x4*)(As_lo + row * P32);
        aH[t].q[0] = qh[half];     aH[t].q[1] = qh[2 + half];
        aL[t].q[0] = ql[half];     aL[t].q[1] = ql[2 + half];
        int col = waveN * 32 + t * 16 + rl;
        const u32x4* ph = (const u32x4*)(Bs_hi + col * P32);
        const u32x4* pl = (const u32x4*)(Bs_lo + col * P32);
        bH[t].q[0] = ph[half];     bH[t].q[1] = ph[2 + half];
        bL[t].q[0] = pl[half];     bL[t].q[1] = pl[2 + half];
    }
#pragma unroll
    for (int tm = 0; tm < 2; ++tm)
#pragma unroll
        for (int tn = 0; tn < 2; ++tn) {
            acc[tm][tn] = __builtin_amdgcn_wmma_f32_16x16x32_bf16(
                false, aL[tm].v, false, bH[tn].v, (short)0, acc[tm][tn], false, false);
            acc[tm][tn] = __builtin_amdgcn_wmma_f32_16x16x32_bf16(
                false, aH[tm].v, false, bL[tn].v, (short)0, acc[tm][tn], false, false);
            acc[tm][tn] = __builtin_amdgcn_wmma_f32_16x16x32_bf16(
                false, aH[tm].v, false, bH[tn].v, (short)0, acc[tm][tn], false, false);
        }
}

// ---------------------------------------------------------------------------
// Kernel 1: per-state scalars.  g2 = 2*exp(gamma_log), k = lambda*x (complex).
// ---------------------------------------------------------------------------
__global__ void lru_scalars(const float* __restrict__ nu_log, const float* __restrict__ theta_log,
                            const float* __restrict__ gamma_log,
                            const float* __restrict__ x_re, const float* __restrict__ x_im,
                            float* __restrict__ g2, float* __restrict__ kr, float* __restrict__ ki) {
    int s = blockIdx.x * blockDim.x + threadIdx.x;
    if (s < DSTATE) {
        float lam_mod = expf(-expf(nu_log[s]));
        float th      = expf(theta_log[s]);
        float lr = lam_mod * cosf(th);
        float li = lam_mod * sinf(th);
        g2[s] = 2.0f * expf(gamma_log[s]);
        kr[s] = lr * x_re[s] - li * x_im[s];
        ki[s] = lr * x_im[s] + li * x_re[s];
    }
}

// ---------------------------------------------------------------------------
// Kernel 2: bias[j] = 2 * sum_s (C_re[j,s]*kr[s] - C_im[j,s]*ki[s]).
// ---------------------------------------------------------------------------
__global__ __launch_bounds__(256) void lru_bias(const float* __restrict__ C_re,
                                                const float* __restrict__ C_im,
                                                const float* __restrict__ kr,
                                                const float* __restrict__ ki,
                                                float* __restrict__ bias) {
    int gwave = (blockIdx.x * blockDim.x + threadIdx.x) >> 5;
    int lane  = threadIdx.x & 31;
    if (gwave >= DOUT) return;
    const float* cr = C_re + (size_t)gwave * DSTATE;
    const float* ci = C_im + (size_t)gwave * DSTATE;
    float acc = 0.0f;
    for (int s = lane; s < DSTATE; s += 32)
        acc += cr[s] * kr[s] - ci[s] * ki[s];
#pragma unroll
    for (int off = 16; off > 0; off >>= 1)
        acc += __shfl_down(acc, off, 32);
    if (lane == 0) bias[gwave] = 2.0f * acc;
}

// ---------------------------------------------------------------------------
// Kernel 3: split u (f32) into bf16 hi/lo planes (one streaming pass).
// ---------------------------------------------------------------------------
__global__ __launch_bounds__(256) void lru_split_u(const float* __restrict__ u,
                                                   unsigned short* __restrict__ uh,
                                                   unsigned short* __restrict__ ul) {
    size_t base = ((size_t)blockIdx.x * 256 + threadIdx.x) * 8;
    f32x4 a = *(const f32x4*)(u + base);
    f32x4 b = *(const f32x4*)(u + base + 4);
    unsigned short h[8], l[8];
#pragma unroll
    for (int j = 0; j < 8; ++j) {
        float x = (j < 4) ? a[j] : b[j - 4];
        h[j] = f2bf(x);
        l[j] = f2bf(x - bf2f(h[j]));
    }
    u32x4 hv, lv;
#pragma unroll
    for (int j = 0; j < 4; ++j) {
        hv[j] = (unsigned)h[2 * j] | ((unsigned)h[2 * j + 1] << 16);
        lv[j] = (unsigned)l[2 * j] | ((unsigned)l[2 * j + 1] << 16);
    }
    *(u32x4*)(uh + base) = hv;
    *(u32x4*)(ul + base) = lv;
}

// ---------------------------------------------------------------------------
// Kernel 4: weight fusion.
//   W[j,i] = D[j,i] + sum_s (g2[s]*C_re[j,s])*B_re[s,i] - (g2[s]*C_im[j,s])*B_im[s,i]
// GEMM M=DOUT, N=DIN, K=2*DSTATE; output written directly as bf16 hi/lo.
// ---------------------------------------------------------------------------
__global__ __launch_bounds__(256) void lru_fuse_w(const float* __restrict__ C_re,
                                                  const float* __restrict__ C_im,
                                                  const float* __restrict__ B_re,
                                                  const float* __restrict__ B_im,
                                                  const float* __restrict__ D,
                                                  const float* __restrict__ g2,
                                                  unsigned short* __restrict__ W_hi,
                                                  unsigned short* __restrict__ W_lo) {
    __shared__ __align__(16) unsigned As_hi[BM * P32];
    __shared__ __align__(16) unsigned As_lo[BM * P32];
    __shared__ __align__(16) unsigned Bs_hi[BN * P32];
    __shared__ __align__(16) unsigned Bs_lo[BN * P32];

    const int tid = threadIdx.x, lane = tid & 31, wave = tid >> 5;
    const int waveM = wave >> 1, waveN = wave & 1;
    const int j0 = blockIdx.y * BM;
    const int i0 = blockIdx.x * BN;

    v8f acc[2][2] = {};

    for (int kc = 0; kc < 2 * DSTATE; kc += BK) {
        const bool  im  = kc >= DSTATE;
        const int   s0  = im ? kc - DSTATE : kc;
        const float sgn = im ? -1.0f : 1.0f;
        const float* Asrc = (im ? C_im : C_re) + (size_t)j0 * DSTATE + s0;
        const float* Bsrc = (im ? B_im : B_re) + (size_t)s0 * DIN + i0;

        stage_tile(Asrc, DSTATE, BM, (unsigned short*)As_hi, (unsigned short*)As_lo,
                   g2 + s0, sgn, tid);
        stage_tile_T(Bsrc, DIN, (unsigned short*)Bs_hi, (unsigned short*)Bs_lo, tid);
        __syncthreads();
        wave_mma(As_hi, As_lo, Bs_hi, Bs_lo, acc, waveM, waveN, lane);
        __syncthreads();
    }

    const int half = lane >> 4, rl = lane & 15;
#pragma unroll
    for (int tm = 0; tm < 2; ++tm)
#pragma unroll
        for (int tn = 0; tn < 2; ++tn) {
            int col     = i0 + waveN * 32 + tn * 16 + rl;
            int rowBase = j0 + waveM * 32 + tm * 16 + 8 * half;
#pragma unroll
            for (int r = 0; r < 8; ++r) {
                size_t idx = (size_t)(rowBase + r) * DIN + col;
                float  wv  = acc[tm][tn][r] + D[idx];
                unsigned short h = f2bf(wv);
                W_hi[idx] = h;
                W_lo[idx] = f2bf(wv - bf2f(h));
            }
        }
}

// ---------------------------------------------------------------------------
// Kernel 5: main GEMM.  y[b,j] = bias[j] + sum_i u[b,i] * W[j,i]
// M=BATCH, N=DOUT, K=DIN.  Pure bf16 feed: double-buffered LDS tiles filled by
// global_load_async_to_lds_b128 (ASYNCcnt) overlapping the WMMA pipeline.
// Whole working set (u 64MB + W 4MB + y 64MB) is L2-resident.
// ---------------------------------------------------------------------------
__global__ __launch_bounds__(256) void lru_main_gemm(const unsigned short* __restrict__ u_hi,
                                                     const unsigned short* __restrict__ u_lo,
                                                     const unsigned short* __restrict__ w_hi,
                                                     const unsigned short* __restrict__ w_lo,
                                                     const float* __restrict__ bias,
                                                     float* __restrict__ y) {
    __shared__ __align__(16) unsigned Auh[2][BM * P32];
    __shared__ __align__(16) unsigned Aul[2][BM * P32];
    __shared__ __align__(16) unsigned Bwh[2][BN * P32];
    __shared__ __align__(16) unsigned Bwl[2][BN * P32];

    const int tid = threadIdx.x, lane = tid & 31, wave = tid >> 5;
    const int waveM = wave >> 1, waveN = wave & 1;
    const int m0 = blockIdx.y * BM;
    const int n0 = blockIdx.x * BN;
    constexpr int NC = DIN / BK;

    v8f acc[2][2] = {};

    // prefetch chunk 0
    async_tile(u_hi + (size_t)m0 * DIN, DIN, Auh[0], BM, tid);
    async_tile(u_lo + (size_t)m0 * DIN, DIN, Aul[0], BM, tid);
    async_tile(w_hi + (size_t)n0 * DIN, DIN, Bwh[0], BN, tid);
    async_tile(w_lo + (size_t)n0 * DIN, DIN, Bwl[0], BN, tid);

    for (int c = 0; c < NC; ++c) {
        wait_async0();          // chunk c landed (this wave's portion)
        __syncthreads();        // all waves' portions landed; compute c-1 fully retired
        if (c + 1 < NC) {       // DMA chunk c+1 while computing chunk c
            int k0 = (c + 1) * BK;
            int nb = (c + 1) & 1;
            async_tile(u_hi + (size_t)m0 * DIN + k0, DIN, Auh[nb], BM, tid);
            async_tile(u_lo + (size_t)m0 * DIN + k0, DIN, Aul[nb], BM, tid);
            async_tile(w_hi + (size_t)n0 * DIN + k0, DIN, Bwh[nb], BN, tid);
            async_tile(w_lo + (size_t)n0 * DIN + k0, DIN, Bwl[nb], BN, tid);
        }
        int b = c & 1;
        wave_mma(Auh[b], Aul[b], Bwh[b], Bwl[b], acc, waveM, waveN, lane);
    }

    const int half = lane >> 4, rl = lane & 15;
#pragma unroll
    for (int tm = 0; tm < 2; ++tm)
#pragma unroll
        for (int tn = 0; tn < 2; ++tn) {
            int col     = n0 + waveN * 32 + tn * 16 + rl;
            float b     = bias[col];
            int rowBase = m0 + waveM * 32 + tm * 16 + 8 * half;
#pragma unroll
            for (int r = 0; r < 8; ++r)
                y[(size_t)(rowBase + r) * DOUT + col] = acc[tm][tn][r] + b;
        }
}

// ---------------------------------------------------------------------------
extern "C" void kernel_launch(void* const* d_in, const int* in_sizes, int n_in,
                              void* d_out, int out_size, void* d_ws, size_t ws_size,
                              hipStream_t stream) {
    const float* u_in      = (const float*)d_in[0];
    const float* x_re      = (const float*)d_in[1];
    const float* x_im      = (const float*)d_in[2];
    const float* nu_log    = (const float*)d_in[3];
    const float* theta_log = (const float*)d_in[4];
    const float* gamma_log = (const float*)d_in[5];
    const float* B_re      = (const float*)d_in[6];
    const float* B_im      = (const float*)d_in[7];
    const float* C_re      = (const float*)d_in[8];
    const float* C_im      = (const float*)d_in[9];
    const float* Dm        = (const float*)d_in[10];

    char* wsb = (char*)d_ws;
    unsigned short* u_hi = (unsigned short*)(wsb);                        // 32 MB
    unsigned short* u_lo = (unsigned short*)(wsb + ((size_t)32 << 20));   // 32 MB
    unsigned short* W_hi = (unsigned short*)(wsb + ((size_t)64 << 20));   //  2 MB
    unsigned short* W_lo = (unsigned short*)(wsb + ((size_t)66 << 20));   //  2 MB
    float* g2   = (float*)(wsb + ((size_t)68 << 20));                     //  8 KB
    float* kr   = g2 + DSTATE;
    float* ki   = kr + DSTATE;
    float* bias = ki + DSTATE;

    // 1) per-state scalars
    lru_scalars<<<(DSTATE + 255) / 256, 256, 0, stream>>>(
        nu_log, theta_log, gamma_log, x_re, x_im, g2, kr, ki);

    // 2) bias matvec (one wave32 per output)
    lru_bias<<<DOUT / 8, 256, 0, stream>>>(C_re, C_im, kr, ki, bias);

    // 3) split u into bf16 hi/lo planes (streaming)
    lru_split_u<<<(BATCH * DIN) / (256 * 8), 256, 0, stream>>>(u_in, u_hi, u_lo);

    // 4) fused weight  W = D + (2g.C_re)*B_re - (2g.C_im)*B_im  (bf16 hi/lo out)
    dim3 gB(DIN / BN, DOUT / BM);                   // 16 x 8
    lru_fuse_w<<<gB, 256, 0, stream>>>(C_re, C_im, B_re, B_im, Dm, g2, W_hi, W_lo);

    // 5) main GEMM  y = u . W^T + bias  (async DMA double-buffered)
    dim3 gC(DOUT / BN, BATCH / BM);                 // 16 x 128
    lru_main_gemm<<<gC, 256, 0, stream>>>(u_hi, u_lo, W_hi, W_lo, bias, (float*)d_out);
}
